// Generator_38259568673048
// MI455X (gfx1250) — matrix operand order repaired and checked by reading, not compiled, not run on hardware
//
#include <hip/hip_runtime.h>
#include <hip/hip_bf16.h>

// ---------------------------------------------------------------------------
// Transformer generator (encoder-decoder greedy decode) for MI455X / gfx1250.
// - All E-dim GEMMs + vocab logits GEMM: v_wmma_f32_16x16x32_f16, fp32 accum.
// - Weights converted+transposed to f16 (N x K) once per launch -> B-fragment
//   is a single contiguous 32B load per lane.
// - Attention Q/K/V tile staging into LDS uses the Tensor Data Mover
//   (tensor_load_to_lds + s_wait_tensorcnt) when the builtin is available.
// ---------------------------------------------------------------------------

typedef __attribute__((ext_vector_type(16))) _Float16 v16h;
typedef __attribute__((ext_vector_type(8)))  float    v8f;

constexpr int kE  = 512;
constexpr int kH  = 8;
constexpr int kM  = 2;
constexpr int kN  = 4;
constexpr int kB  = 16;
constexpr int kS  = 32;
constexpr int kV  = 10000;
constexpr int kDh = 64;
constexpr int kT  = kB * kS;     // 512 tokens per activation matrix
constexpr size_t kEE = (size_t)kE * kE;

// ---------------------------------------------------------------------------
// Tensor Data Mover: 2D tile (rows x cols fp32, row stride in elements) from
// global memory into LDS. D# layout per CDNA5 ISA ch.8 (group0 128b, group1
// 256b). Issued by one wave; completion via TENSORcnt.
// ---------------------------------------------------------------------------
#if __has_builtin(__builtin_amdgcn_tensor_load_to_lds)
#define HAVE_TDM 1
typedef unsigned int u32x4 __attribute__((ext_vector_type(4)));
typedef int          i32x4 __attribute__((ext_vector_type(4)));
typedef int          i32x8 __attribute__((ext_vector_type(8)));

__device__ __forceinline__ void tdm_load_2d_f32(const float* gsrc, unsigned ldsoff,
                                                int rows, int cols, int strideElems)
{
    unsigned long long ga = (unsigned long long)(uintptr_t)gsrc;
    u32x4 g0;
    g0[0] = 1u;                                          // count=1, user D#, no gather
    g0[1] = ldsoff;                                      // lds_addr (bytes)
    g0[2] = (unsigned)(ga & 0xffffffffu);                // global_addr[31:0]
    g0[3] = (unsigned)((ga >> 32) & 0x01ffffffu)         // global_addr[56:32]
          | (2u << 30);                                  // type=2 ("image")
    const unsigned td0 = 0x40000000u;                    // huge tensor dims: tile is
    const unsigned td1 = 0x40000000u;                    // fully in-bounds, no OOB clamp
    i32x8 g1;
    g1[0] = 0x00020000;                                  // data_size=2 (4B); mask/flags=0
    g1[1] = (int)((td0 & 0xffffu) << 16);                // tensor_dim0[15:0] @ bits63:48
    g1[2] = (int)((td0 >> 16) | ((td1 & 0xffffu) << 16));// dim0[31:16] | dim1[15:0]
    g1[3] = (int)((td1 >> 16) | ((unsigned)cols << 16)); // dim1[31:16] | tile_dim0
    g1[4] = rows & 0xffff;                               // tile_dim1 | tile_dim2=0 (2D)
    g1[5] = strideElems;                                 // tensor_dim0_stride[31:0]
    g1[6] = 0;                                           // stride[47:32] | dim1_stride lo
    g1[7] = 0;
    i32x4 gz = {0, 0, 0, 0};
#if __clang_major__ >= 23
    i32x8 gz8 = {0, 0, 0, 0, 0, 0, 0, 0};
    __builtin_amdgcn_tensor_load_to_lds(g0, g1, gz, gz, gz8, 0);
#else
    __builtin_amdgcn_tensor_load_to_lds(g0, g1, gz, gz, 0);
#endif
}

__device__ __forceinline__ void tdm_wait()
{
#if __has_builtin(__builtin_amdgcn_s_wait_tensorcnt)
    __builtin_amdgcn_s_wait_tensorcnt(0);
#else
    asm volatile("s_wait_tensorcnt 0x0" ::: "memory");
#endif
}
#endif // HAVE_TDM

// ---------------------------------------------------------------------------
// WMMA GEMM: out[T x N] = act(A[T x K] @ Wt^T + bias), Wt stored N x K (f16).
// Each wave computes a 16 x (16*NT) strip; A fp32 -> f16 in-register.
// ---------------------------------------------------------------------------
template<int NT>
__global__ void gemm_wmma_kernel(const float* __restrict__ A, int lda,
                                 const _Float16* __restrict__ Wt,
                                 const float* __restrict__ bias,
                                 float* __restrict__ out, int ldo,
                                 int T, int K, int N, int relu)
{
    const int lane = threadIdx.x & 31;
    const int wave = threadIdx.x >> 5;
    const int wavesPerBlock = blockDim.x >> 5;
    const int nStrips = N / (16 * NT);
    const int mTiles  = T / 16;
    const int strip = blockIdx.x * wavesPerBlock + wave;
    if (strip >= mTiles * nStrips) return;        // wave-uniform: EXEC stays full
    const int m0 = (strip / nStrips) * 16;
    const int n0 = (strip % nStrips) * (16 * NT);

    const int row   = m0 + (lane & 15);
    const int kA    = (lane >> 4) * 16;
    const int kBofs = (lane >> 4) * 16;
    const int nlane = lane & 15;

    const v8f vzero = {0.f, 0.f, 0.f, 0.f, 0.f, 0.f, 0.f, 0.f};
    v8f acc[NT];
    #pragma unroll
    for (int t = 0; t < NT; ++t) acc[t] = vzero;

    for (int k0 = 0; k0 < K; k0 += 32) {
        const float* ap = A + (size_t)row * lda + k0 + kA;
        v16h a;
        #pragma unroll
        for (int j = 0; j < 8; ++j) a[j]     = (_Float16)ap[j];
        #pragma unroll
        for (int j = 0; j < 8; ++j) a[8 + j] = (_Float16)ap[8 + j];
        #pragma unroll
        for (int t = 0; t < NT; ++t) {
            const int n = n0 + t * 16 + nlane;
            v16h b = *(const v16h*)(Wt + (size_t)n * K + k0 + kBofs);
            acc[t] = __builtin_amdgcn_wmma_f32_16x16x32_f16(
                         false, a, false, b, (short)0, acc[t], false, false);
        }
    }

    const int rbase = m0 + (lane >> 4) * 8;
    #pragma unroll
    for (int t = 0; t < NT; ++t) {
        const int n = n0 + t * 16 + nlane;
        const float bv = bias ? bias[n] : 0.0f;
        #pragma unroll
        for (int r = 0; r < 8; ++r) {
            float v = acc[t][r] + bv;
            if (relu) v = fmaxf(v, 0.0f);
            out[(size_t)(rbase + r) * ldo + n] = v;
        }
    }
}

// ---------------------------------------------------------------------------
// Weight convert + transpose: dst[n*K + k] = (f16) src[k*N + n]
// ---------------------------------------------------------------------------
__global__ void convT_kernel(const float* __restrict__ src, _Float16* __restrict__ dst,
                             int K, int N)
{
    size_t i = (size_t)blockIdx.x * 256 + threadIdx.x;
    if (i >= (size_t)K * N) return;
    int k = (int)(i % K);
    int n = (int)(i / K);
    dst[i] = (_Float16)src[(size_t)k * N + n];
}

// ---------------------------------------------------------------------------
// Attention for one (batch, head) per block. S=32, Dh=64. 128 threads.
// Q/K/V tiles staged into LDS by the Tensor Data Mover when available.
// ---------------------------------------------------------------------------
__global__ void attention_kernel(const float* __restrict__ Q, int ldq,
                                 const float* __restrict__ Kp, int ldk,
                                 const float* __restrict__ Vp, int ldv,
                                 float* __restrict__ O, int ldo, float scale)
{
    const int b = blockIdx.x / kH;
    const int h = blockIdx.x % kH;
    const int tid = threadIdx.x;

    __shared__ float sQ[kS * kDh];
    __shared__ float sK[kS * kDh];
    __shared__ float sV[kS * kDh];
    __shared__ float sAtt[kS * kS];

    const float* gQ = Q  + (size_t)(b * kS) * ldq + h * kDh;
    const float* gK = Kp + (size_t)(b * kS) * ldk + h * kDh;
    const float* gV = Vp + (size_t)(b * kS) * ldv + h * kDh;

#if defined(HAVE_TDM)
    if (tid < 32) {   // wave 0 owns the three tile DMAs (TENSORcnt is per-wave)
        tdm_load_2d_f32(gQ, (unsigned)(uintptr_t)(void*)sQ, kS, kDh, ldq);
        tdm_load_2d_f32(gK, (unsigned)(uintptr_t)(void*)sK, kS, kDh, ldk);
        tdm_load_2d_f32(gV, (unsigned)(uintptr_t)(void*)sV, kS, kDh, ldv);
        tdm_wait();
    }
    __syncthreads();
#else
    for (int q = tid; q < (kS * kDh) / 4; q += 128) {
        int s = q >> 4, dq = q & 15;                    // 16 float4 per 64-wide row
        *(float4*)&sQ[s * kDh + dq * 4] = *(const float4*)(gQ + (size_t)s * ldq + dq * 4);
        *(float4*)&sK[s * kDh + dq * 4] = *(const float4*)(gK + (size_t)s * ldk + dq * 4);
        *(float4*)&sV[s * kDh + dq * 4] = *(const float4*)(gV + (size_t)s * ldv + dq * 4);
    }
    __syncthreads();
#endif

    for (int i = tid; i < kS * kS; i += 128) {
        int q = i / kS, k = i % kS;
        float acc = 0.f;
        #pragma unroll 8
        for (int d = 0; d < kDh; ++d) acc += sQ[q * kDh + d] * sK[k * kDh + d];
        sAtt[i] = acc * scale;
    }
    __syncthreads();

    if (tid < kS) {
        float mx = -3.4e38f;
        for (int k = 0; k < kS; ++k) mx = fmaxf(mx, sAtt[tid * kS + k]);
        float sum = 0.f;
        for (int k = 0; k < kS; ++k) {
            float e = __expf(sAtt[tid * kS + k] - mx);
            sAtt[tid * kS + k] = e;
            sum += e;
        }
        float inv = 1.f / sum;
        for (int k = 0; k < kS; ++k) sAtt[tid * kS + k] *= inv;
    }
    __syncthreads();

    for (int i = tid; i < kS * kDh; i += 128) {
        int q = i / kDh, d = i % kDh;
        float acc = 0.f;
        #pragma unroll 8
        for (int k = 0; k < kS; ++k) acc += sAtt[q * kS + k] * sV[k * kDh + d];
        O[(size_t)(b * kS + q) * ldo + h * kDh + d] = acc;
    }
}

// ---------------------------------------------------------------------------
// LayerNorm(X + R): gain=gb[0..E), bias=gb[E..2E). One row per block.
// ---------------------------------------------------------------------------
__global__ void ln_kernel(const float* __restrict__ X, const float* __restrict__ R,
                          const float* __restrict__ gb, float* __restrict__ out)
{
    const int row = blockIdx.x;
    const int tid = threadIdx.x;
    const float* x = X + (size_t)row * kE;
    const float* r = R + (size_t)row * kE;

    float v0 = x[tid]       + r[tid];
    float v1 = x[tid + 256] + r[tid + 256];

    __shared__ float ssum[256];
    __shared__ float ssq[256];
    ssum[tid] = v0 + v1;
    ssq[tid]  = v0 * v0 + v1 * v1;
    __syncthreads();
    for (int s = 128; s > 0; s >>= 1) {
        if (tid < s) { ssum[tid] += ssum[tid + s]; ssq[tid] += ssq[tid + s]; }
        __syncthreads();
    }
    const float mean = ssum[0] * (1.0f / kE);
    const float var  = ssq[0] * (1.0f / kE) - mean * mean;
    const float rs   = rsqrtf(var + 1e-5f);

    out[(size_t)row * kE + tid]       = (v0 - mean) * rs * gb[tid]       + gb[kE + tid];
    out[(size_t)row * kE + tid + 256] = (v1 - mean) * rs * gb[tid + 256] + gb[kE + tid + 256];
}

// ---------------------------------------------------------------------------
// Softmax over V logits per batch row + first-occurrence argmax.
// ---------------------------------------------------------------------------
__global__ void softmax_argmax_kernel(const float* __restrict__ logits,
                                      float* __restrict__ probs,
                                      int* __restrict__ tok_ids,
                                      float* __restrict__ tok_out)
{
    const int b = blockIdx.x;
    const int tid = threadIdx.x;
    const float* x = logits + (size_t)b * kV;

    float mx = -3.4e38f; int arg = 0x7fffffff;
    for (int i = tid; i < kV; i += 256) {
        float v = x[i];
        if (v > mx) { mx = v; arg = i; }
    }
    __shared__ float smax[256];
    __shared__ int   sarg[256];
    __shared__ float ssum[256];
    smax[tid] = mx; sarg[tid] = arg;
    __syncthreads();
    for (int s = 128; s > 0; s >>= 1) {
        if (tid < s) {
            if (smax[tid + s] > smax[tid] ||
                (smax[tid + s] == smax[tid] && sarg[tid + s] < sarg[tid])) {
                smax[tid] = smax[tid + s];
                sarg[tid] = sarg[tid + s];
            }
        }
        __syncthreads();
    }
    const float gmax = smax[0];
    const int   garg = sarg[0];

    float lsum = 0.f;
    for (int i = tid; i < kV; i += 256) lsum += __expf(x[i] - gmax);
    ssum[tid] = lsum;
    __syncthreads();
    for (int s = 128; s > 0; s >>= 1) {
        if (tid < s) ssum[tid] += ssum[tid + s];
        __syncthreads();
    }
    const float inv = 1.f / ssum[0];
    for (int i = tid; i < kV; i += 256)
        probs[(size_t)b * (kS - 1) * kV + i] = __expf(x[i] - gmax) * inv;

    if (tid == 0) {
        tok_ids[b] = garg;
        if (tok_out) tok_out[b] = (float)garg;
    }
}

__global__ void writeback_kernel(float* __restrict__ Ycur, const float* __restrict__ embed,
                                 const int* __restrict__ tok_ids, int tok)
{
    const int b = blockIdx.x;
    const int e = threadIdx.x;           // 512 threads
    const int id = tok_ids[b];
    Ycur[((size_t)b * kS + tok) * kE + e] = embed[(size_t)id * kE + e];
}

__global__ void init_y0_kernel(float* __restrict__ Y, const float* __restrict__ embed)
{
    size_t i = (size_t)blockIdx.x * 256 + threadIdx.x;   // over B*S*E
    int e = (int)(i % kE);
    int s = (int)((i / kE) % kS);
    int tok = (s == 0) ? 1 : 0;                          // START vs PAD
    Y[i] = embed[(size_t)tok * kE + e];
}

__global__ void pe_kernel(float* __restrict__ pe)
{
    int i = blockIdx.x * 256 + threadIdx.x;              // over S*E
    int e = i % kE;
    int s = i / kE;
    int pair = e >> 1;
    float div = __expf((float)(2 * pair) * (-9.210340371976184f / (float)kE));
    float ang = (float)s * div;
    pe[i] = (e & 1) ? __cosf(ang) : __sinf(ang);
}

__global__ void add_pe_kernel(const float* __restrict__ Y, const float* __restrict__ pe,
                              float* __restrict__ out)
{
    size_t i = (size_t)blockIdx.x * 256 + threadIdx.x;   // over B*S*E
    out[i] = Y[i] + pe[i % (size_t)(kS * kE)];
}

// ---------------------------------------------------------------------------
// Host orchestration
// ---------------------------------------------------------------------------
extern "C" void kernel_launch(void* const* d_in, const int* in_sizes, int n_in,
                              void* d_out, int out_size, void* d_ws, size_t ws_size,
                              hipStream_t stream)
{
    (void)in_sizes; (void)n_in; (void)out_size; (void)ws_size;

    const float* noise      = (const float*)d_in[0];
    const float* in_attn_W  = (const float*)d_in[1];
    const float* in_attn_b  = (const float*)d_in[2];
    const float* in_ffn_W   = (const float*)d_in[3];
    const float* in_ffn_b   = (const float*)d_in[4];
    const float* in_ln      = (const float*)d_in[5];
    const float* out_attn_W = (const float*)d_in[6];
    const float* out_attn_b = (const float*)d_in[7];
    const float* out_ffn_W  = (const float*)d_in[8];
    const float* out_ffn_b  = (const float*)d_in[9];
    const float* out_ln     = (const float*)d_in[10];
    const float* embed      = (const float*)d_in[11];
    const float* soft_W     = (const float*)d_in[12];
    const float* soft_b     = (const float*)d_in[13];
    float* outp = (float*)d_out;

    // ---- workspace carve-out (256B aligned) ----
    char* wsb = (char*)d_ws;
    size_t off = 0;
    auto alloc = [&](size_t bytes) -> char* {
        char* p = wsb + off;
        off = (off + bytes + 255) & ~(size_t)255;
        return p;
    };
    _Float16* encQKV = (_Float16*)alloc(kM * 3 * kEE * 2);
    _Float16* encWo  = (_Float16*)alloc(kM * kEE * 2);
    _Float16* encF   = (_Float16*)alloc(kM * 2 * kEE * 2);
    _Float16* decQKV = (_Float16*)alloc(kN * 3 * kEE * 2);
    _Float16* decWo  = (_Float16*)alloc(kN * kEE * 2);
    _Float16* decCQ  = (_Float16*)alloc(kN * kEE * 2);
    _Float16* decCKV = (_Float16*)alloc(kN * 2 * kEE * 2);
    _Float16* decCWo = (_Float16*)alloc(kN * kEE * 2);
    _Float16* decF   = (_Float16*)alloc(kN * 2 * kEE * 2);
    _Float16* softWt = (_Float16*)alloc((size_t)kV * kE * 2);
    float* Z      = (float*)alloc((size_t)kT * kE * 4);
    float* Ycur   = (float*)alloc((size_t)kT * kE * 4);
    float* Ybuf   = (float*)alloc((size_t)kT * kE * 4);
    float* qkvb   = (float*)alloc((size_t)kT * 3 * kE * 4);
    float* attb   = (float*)alloc((size_t)kT * kE * 4);
    float* tmpb   = (float*)alloc((size_t)kT * kE * 4);
    float* hbuf   = (float*)alloc((size_t)kT * kE * 4);
    float* kvz    = (float*)alloc((size_t)kN * kT * 2 * kE * 4);
    float* logits = (float*)alloc((size_t)kB * kV * 4);
    float* peb    = (float*)alloc((size_t)kS * kE * 4);
    int*   tokids = (int*)alloc(kB * sizeof(int));

    // ---- helpers ----
    auto conv = [&](const float* src, _Float16* dst, int K, int N) {
        size_t total = (size_t)K * N;
        int blocks = (int)((total + 255) / 256);
        convT_kernel<<<blocks, 256, 0, stream>>>(src, dst, K, N);
    };
    auto gemm4 = [&](const float* A, int lda, const _Float16* Wt, const float* bias,
                     float* O, int ldo, int T, int K, int N, int relu) {
        int strips = (T / 16) * (N / 64);
        int blocks = (strips + 7) / 8;
        gemm_wmma_kernel<4><<<blocks, 256, 0, stream>>>(A, lda, Wt, bias, O, ldo, T, K, N, relu);
    };
    auto attn = [&](const float* Q, int ldq, const float* Kp, int ldk,
                    const float* Vp, int ldv, float* O, int ldo) {
        attention_kernel<<<kB * kH, 128, 0, stream>>>(Q, ldq, Kp, ldk, Vp, ldv, O, ldo, 0.125f);
    };
    auto lnr = [&](const float* X, const float* R, const float* gb, float* O) {
        ln_kernel<<<kT, 256, 0, stream>>>(X, R, gb, O);
    };

    // ---- one-time weight conversion + transpose to f16 N x K ----
    for (int m = 0; m < kM; ++m) {
        for (int i = 0; i < 3; ++i)
            conv(in_attn_W + (size_t)(m * 4 + i) * kEE, encQKV + (size_t)(m * 3 + i) * kEE, kE, kE);
        conv(in_attn_W + (size_t)(m * 4 + 3) * kEE, encWo + (size_t)m * kEE, kE, kE);
        for (int i = 0; i < 2; ++i)
            conv(in_ffn_W + (size_t)(m * 2 + i) * kEE, encF + (size_t)(m * 2 + i) * kEE, kE, kE);
    }
    for (int n = 0; n < kN; ++n) {
        for (int i = 0; i < 3; ++i)
            conv(out_attn_W + (size_t)(n * 8 + i) * kEE, decQKV + (size_t)(n * 3 + i) * kEE, kE, kE);
        conv(out_attn_W + (size_t)(n * 8 + 3) * kEE, decWo + (size_t)n * kEE, kE, kE);
        conv(out_attn_W + (size_t)(n * 8 + 4) * kEE, decCQ + (size_t)n * kEE, kE, kE);
        for (int i = 0; i < 2; ++i)
            conv(out_attn_W + (size_t)(n * 8 + 5 + i) * kEE, decCKV + (size_t)(n * 2 + i) * kEE, kE, kE);
        conv(out_attn_W + (size_t)(n * 8 + 7) * kEE, decCWo + (size_t)n * kEE, kE, kE);
        for (int i = 0; i < 2; ++i)
            conv(out_ffn_W + (size_t)(n * 2 + i) * kEE, decF + (size_t)(n * 2 + i) * kEE, kE, kE);
    }
    conv(soft_W, softWt, kE, kV);

    // ---- init: positional encodings, Z = noise, Y0 = PAD/START embeddings ----
    pe_kernel<<<(kS * kE) / 256, 256, 0, stream>>>(peb);
    hipMemcpyAsync(Z, noise, (size_t)kT * kE * 4, hipMemcpyDeviceToDevice, stream);
    init_y0_kernel<<<(kT * kE) / 256, 256, 0, stream>>>(Ycur, embed);

    // ---- encoder (M blocks), Z fixed thereafter ----
    for (int m = 0; m < kM; ++m) {
        const float* ab = in_attn_b + (size_t)m * 4 * kE;
        gemm4(Z, kE, encQKV + (size_t)m * 3 * kEE, ab, qkvb, 3 * kE, kT, kE, 3 * kE, 0);
        attn(qkvb, 3 * kE, qkvb + kE, 3 * kE, qkvb + 2 * kE, 3 * kE, attb, kE);
        gemm4(attb, kE, encWo + (size_t)m * kEE, ab + 3 * kE, tmpb, kE, kT, kE, kE, 0);
        lnr(Z, tmpb, in_ln + (size_t)(m * 2 + 0) * 2 * kE, Z);
        const float* fb = in_ffn_b + (size_t)m * 2 * kE;
        gemm4(Z, kE, encF + (size_t)(m * 2 + 0) * kEE, fb, hbuf, kE, kT, kE, kE, 1);
        gemm4(hbuf, kE, encF + (size_t)(m * 2 + 1) * kEE, fb + kE, tmpb, kE, kT, kE, kE, 0);
        lnr(Z, tmpb, in_ln + (size_t)(m * 2 + 1) * 2 * kE, Z);
    }

    // ---- cross-attention K,V of Z are loop-invariant: precompute per block ----
    for (int n = 0; n < kN; ++n) {
        gemm4(Z, kE, decCKV + (size_t)n * 2 * kEE,
              out_attn_b + (size_t)n * 8 * kE + 5 * kE,
              kvz + (size_t)n * kT * 2 * kE, 2 * kE, kT, kE, 2 * kE, 0);
    }

    // ---- greedy decode: 31 sequential steps ----
    for (int t = 1; t < kS; ++t) {
        add_pe_kernel<<<(kT * kE) / 256, 256, 0, stream>>>(Ycur, peb, Ybuf);
        for (int n = 0; n < kN; ++n) {
            const float* ab  = out_attn_b + (size_t)n * 8 * kE;
            const float* lnp = out_ln + (size_t)n * 3 * 2 * kE;
            // self-attention
            gemm4(Ybuf, kE, decQKV + (size_t)n * 3 * kEE, ab, qkvb, 3 * kE, kT, kE, 3 * kE, 0);
            attn(qkvb, 3 * kE, qkvb + kE, 3 * kE, qkvb + 2 * kE, 3 * kE, attb, kE);
            gemm4(attb, kE, decWo + (size_t)n * kEE, ab + 3 * kE, tmpb, kE, kT, kE, kE, 0);
            lnr(Ybuf, tmpb, lnp + 0 * 2 * kE, Ybuf);
            // cross-attention (K,V precomputed from Z)
            gemm4(Ybuf, kE, decCQ + (size_t)n * kEE, ab + 4 * kE, qkvb, kE, kT, kE, kE, 0);
            attn(qkvb, kE,
                 kvz + (size_t)n * kT * 2 * kE, 2 * kE,
                 kvz + (size_t)n * kT * 2 * kE + kE, 2 * kE, attb, kE);
            gemm4(attb, kE, decCWo + (size_t)n * kEE, ab + 7 * kE, tmpb, kE, kT, kE, kE, 0);
            lnr(Ybuf, tmpb, lnp + 1 * 2 * kE, Ybuf);
            // FFN
            const float* fb = out_ffn_b + (size_t)n * 2 * kE;
            gemm4(Ybuf, kE, decF + (size_t)(n * 2 + 0) * kEE, fb, hbuf, kE, kT, kE, kE, 1);
            gemm4(hbuf, kE, decF + (size_t)(n * 2 + 1) * kEE, fb + kE, tmpb, kE, kT, kE, kE, 0);
            lnr(Ybuf, tmpb, lnp + 2 * 2 * kE, Ybuf);
        }
        // logits for token t: rows Ybuf[b*S + t]; 16 x 10000, NT=5 (10000 = 80*125)
        {
            int strips = (16 / 16) * (kV / 80);      // 125
            int blocks = (strips + 7) / 8;
            gemm_wmma_kernel<5><<<blocks, 256, 0, stream>>>(
                Ybuf + (size_t)t * kE, kS * kE, softWt, soft_b,
                logits, kV, 16, kE, kV, 0);
        }
        softmax_argmax_kernel<<<kB, 256, 0, stream>>>(
            logits, outp + (size_t)(t - 1) * kV, tokids,
            (t == 1) ? (outp + (size_t)kB * (kS - 1) * kV) : nullptr);
        writeback_kernel<<<kB, 512, 0, stream>>>(Ycur, embed, tokids, t);
    }
}
